// GELU75_64458869179056
// MI455X (gfx1250) — compile-verified, matrix operand order checked by reading, not run on hardware
//
#include <hip/hip_runtime.h>
#include <hip/hip_bf16.h>
#include <math.h>

// ---------------------------------------------------------------------------
// Types for CDNA5 builtins
// ---------------------------------------------------------------------------
typedef __attribute__((ext_vector_type(16))) __bf16 v16bf;
typedef __attribute__((ext_vector_type(8)))  float  v8f;
typedef __attribute__((ext_vector_type(4)))  unsigned int v4u;
typedef __attribute__((ext_vector_type(8)))  int    v8i;
typedef __attribute__((ext_vector_type(4)))  int    v4i;

#define B_  8
#define T_  4096
#define D_  1024
#define K_  64        // chunk length (tokens per Gram block)
#define NCHUNK (T_ / K_)
#define EPS_ 1e-5f

// LDS layout (floats)
#define XC_OFF    0                    // 64 x 1024 chunk of x (f32)
#define XC_SZ     (K_ * D_)
#define S_OFF     (XC_OFF + XC_SZ)     // 80 x 80 Gram
#define S_STRIDE  80
#define S_SZ      (80 * 80)
#define E_OFF     (S_OFF + S_SZ)       // running EMA vector
#define EG_OFF    (E_OFF + D_)         // ema_global
#define EO_OFF    (EG_OFF + D_)        // normalized ema_out
#define P_OFF     (EO_OFF + D_)        // p_j = x_j . ema_r
#define SL_OFF    (P_OFF + K_)         // surp_l per token of chunk
#define WGT_OFF   (SL_OFF + K_)        // EMA tail weights
#define RED_OFF   (WGT_OFF + K_)       // 32 reduction scratch
#define AS_OFF    (RED_OFF + 32)       // scalar a = ||ema_r||^2
#define SMEM_FLOATS (AS_OFF + 8)
#define SMEM_BYTES  (SMEM_FLOATS * sizeof(float))

__device__ __forceinline__ float gelu_f(float x) {
    float x3 = x * x * x;
    return 0.5f * x * (1.0f + tanhf(0.7978845608028654f * (x + 0.044715f * x3)));
}

// Convert 8 consecutive f32 (16B-aligned) into fragment slots [base, base+8)
__device__ __forceinline__ void cvt8(const float* p, v16bf& f, const int base) {
    float4 a = *(const float4*)(p);
    float4 b = *(const float4*)(p + 4);
    f[base + 0] = (__bf16)a.x; f[base + 1] = (__bf16)a.y;
    f[base + 2] = (__bf16)a.z; f[base + 3] = (__bf16)a.w;
    f[base + 4] = (__bf16)b.x; f[base + 5] = (__bf16)b.y;
    f[base + 6] = (__bf16)b.z; f[base + 7] = (__bf16)b.w;
}

// Branch-free fragment from the 64x1024 f32 chunk.
// bf16 16x32 fragment layout (ISA 7.12.2): lane -> M = lane&15,
// K in {kb..kb+7} U {kb+16..kb+23}, kb = (lane&16) ? 8 : 0.
// Both runs are contiguous -> 4x ds_load_b128 + cvt.
__device__ __forceinline__ v16bf load_frag_x(const float* __restrict__ Xc,
                                             int rowBase, int kBase, int lane) {
    const int m  = lane & 15;
    const int kb = (lane & 16) ? 8 : 0;
    const float* rp = Xc + (rowBase + m) * D_ + kBase + kb;
    v16bf f;
    cvt8(rp,      f, 0);
    cvt8(rp + 16, f, 8);
    return f;
}

// Augmented tile rows 64..79 = [E; ema_g; zeros]. Single hoisted per-lane
// pointer select; no per-element branching.
__device__ __forceinline__ v16bf load_frag_aug(const float* __restrict__ E,
                                               const float* __restrict__ Eg,
                                               int kBase, int lane) {
    const int m  = lane & 15;
    const int kb = (lane & 16) ? 8 : 0;
    const int k0 = kBase + kb;
    v16bf f;
    const float* src = (m == 0) ? E : ((m == 1) ? Eg : (const float*)0);
    if (src) {
        cvt8(src + k0,      f, 0);
        cvt8(src + k0 + 16, f, 8);
    } else {
#pragma unroll
        for (int i = 0; i < 16; ++i) f[i] = (__bf16)0.0f;
    }
    return f;
}

// ---------------------------------------------------------------------------
// Kernel A: one workgroup per batch; streams 64 chunks of 64 tokens.
// Per chunk: TDM load -> WMMA Gram -> O(K^2) scalar scan -> gate_raw.
// ---------------------------------------------------------------------------
__global__ void __launch_bounds__(256)
gate_kernel(const float* __restrict__ x,
            const float* __restrict__ p_ldl, const float* __restrict__ p_ltau,
            const float* __restrict__ p_lsg, const float* __restrict__ p_lsl,
            const float* __restrict__ p_lwg, const float* __restrict__ p_lwl,
            const float* __restrict__ emaG_g, const float* __restrict__ emaO_g,
            float* __restrict__ gate) {
    extern __shared__ float smem[];
    float* Xc    = smem + XC_OFF;
    float* S     = smem + S_OFF;
    float* E     = smem + E_OFF;
    float* Eg    = smem + EG_OFF;
    float* eo    = smem + EO_OFF;
    float* p     = smem + P_OFF;
    float* surpL = smem + SL_OFF;
    float* wgt   = smem + WGT_OFF;
    float* red   = smem + RED_OFF;
    float* aS    = smem + AS_OFF;

    const int tid  = threadIdx.x;
    const int lane = tid & 31;
    const int w    = tid >> 5;
    const int b    = blockIdx.x;

    // scalar parameters (per reference)
    const float d_l   = 1.0f / (1.0f + expf(-p_ldl[0]));
    const float tau   = expf(p_ltau[0]);
    const float sig_g = log1pf(expf(p_lsg[0]));
    const float sig_l = log1pf(expf(p_lsl[0]));
    const float w_g   = log1pf(expf(p_lwg[0]));
    const float w_l   = log1pf(expf(p_lwl[0]));
    const float dpow64 = powf(d_l, (float)K_);

    // ---- setup: copy ema vectors, compute their norms -----------------------
    float so = 0.0f, sg2 = 0.0f;
    for (int i = tid; i < D_; i += 256) {
        float vo = emaO_g[i]; so  += vo * vo;
        float vg = emaG_g[i]; sg2 += vg * vg;
        Eg[i] = vg; E[i] = vg;
    }
#pragma unroll
    for (int off = 16; off; off >>= 1) {
        so  += __shfl_xor(so,  off, 32);
        sg2 += __shfl_xor(sg2, off, 32);
    }
    if (lane == 0) { red[w] = so; red[8 + w] = sg2; }
    __syncthreads();
    if (tid == 0) {
        float a = 0.f, g2 = 0.f;
        for (int i = 0; i < 8; ++i) { a += red[i]; g2 += red[8 + i]; }
        red[16] = sqrtf(a);   // ||ema_out||
        red[17] = sqrtf(g2);  // ||ema_global||
    }
    __syncthreads();
    const float normO = red[16];
    const float normG = red[17];
    for (int i = tid; i < D_; i += 256) eo[i] = emaO_g[i] / fmaxf(normO, 1e-12f);
    if (tid < K_) wgt[tid] = (1.0f - d_l) * powf(d_l, (float)(K_ - 1 - tid));
    __syncthreads();

    // ---- chunk loop ---------------------------------------------------------
    for (int c = 0; c < NCHUNK; ++c) {
        __syncthreads();  // previous chunk fully consumed before overwriting Xc

        const float* xb = x + (size_t)b * T_ * D_ + (size_t)c * K_ * D_;

        // (1) TDM: async tensor load 64x1024 f32 tile into LDS
#if __has_builtin(__builtin_amdgcn_tensor_load_to_lds)
        if (tid < 32) {
            unsigned long long ga = (unsigned long long)(uintptr_t)xb;
            unsigned ldsA = (unsigned)(uintptr_t)Xc;   // low 32 bits = LDS offset
            v4u g0 = { 1u,                              // count=1, not restore
                       ldsA,                            // lds_addr
                       (unsigned)(ga & 0xFFFFFFFFull),  // global_addr[31:0]
                       (unsigned)((ga >> 32) & 0x1FFFFFFull) | (2u << 30) }; // [56:32] | type=2
            v8i g1 = { (int)(2u << 16),          // data_size = 4B
                       (int)(1024u << 16),       // tensor_dim0 = 1024 (lo16 at bit48)
                       (int)(64u << 16),         // tensor_dim1 = 64  (lo16 at bit80)
                       (int)(1024u << 16),       // tile_dim0 = 1024
                       (int)64,                  // tile_dim1 = 64
                       (int)1024,                // tensor_dim0_stride = 1024
                       0, 0 };
            v4i g2 = { 0, 0, 0, 0 };
            v4i g3 = { 0, 0, 0, 0 };
#if defined(__clang_major__) && (__clang_major__ >= 23)
            v8i gz = { 0, 0, 0, 0, 0, 0, 0, 0 };
            __builtin_amdgcn_tensor_load_to_lds(g0, g1, g2, g3, gz, 0);
#else
            __builtin_amdgcn_tensor_load_to_lds(g0, g1, g2, g3, 0);
#endif
            __builtin_amdgcn_s_wait_tensorcnt(0);
        }
#else
        for (int i = tid; i < XC_SZ; i += 256) Xc[i] = xb[i];
#endif
        __syncthreads();

        // (2) Gram S = R R^T via v_wmma_f32_16x16x32_bf16.
        // R = [X_chunk(64 rows); E; ema_g; 0-pad] -> 5x5 tile grid; we need
        // the 4x4 X-block, tile-row 4 vs X (E/ema_g dot products), and (4,4).
        // 21 tiles distributed over 8 waves; tile kind hoisted out of k-loop.
        for (int idx = w; idx < 21; idx += 8) {
            int ti, tj;
            if (idx < 16)      { ti = idx >> 2; tj = idx & 3; }
            else if (idx < 20) { ti = 4;        tj = idx - 16; }
            else               { ti = 4;        tj = 4; }
            v8f acc = {0.f, 0.f, 0.f, 0.f, 0.f, 0.f, 0.f, 0.f};
            if (ti < 4) {                       // X x X
#pragma unroll 2
                for (int kt = 0; kt < D_; kt += 32) {
                    v16bf af = load_frag_x(Xc, ti * 16, kt, lane);
                    v16bf bf = load_frag_x(Xc, tj * 16, kt, lane);
                    acc = __builtin_amdgcn_wmma_f32_16x16x32_bf16(
                        false, af, false, bf, (short)0, acc, false, false);
                }
            } else if (tj < 4) {                // Aug x X
#pragma unroll 2
                for (int kt = 0; kt < D_; kt += 32) {
                    v16bf af = load_frag_aug(E, Eg, kt, lane);
                    v16bf bf = load_frag_x(Xc, tj * 16, kt, lane);
                    acc = __builtin_amdgcn_wmma_f32_16x16x32_bf16(
                        false, af, false, bf, (short)0, acc, false, false);
                }
            } else {                            // Aug x Aug
#pragma unroll 2
                for (int kt = 0; kt < D_; kt += 32) {
                    v16bf af = load_frag_aug(E, Eg, kt, lane);
                    v16bf bf = load_frag_aug(E, Eg, kt, lane);
                    acc = __builtin_amdgcn_wmma_f32_16x16x32_bf16(
                        false, af, false, bf, (short)0, acc, false, false);
                }
            }
            int ro = ti * 16 + ((lane & 16) ? 8 : 0);
            int co = tj * 16 + (lane & 15);
#pragma unroll
            for (int v = 0; v < 8; ++v) S[(ro + v) * S_STRIDE + co] = acc[v];
        }
        __syncthreads();

        // (3) in-chunk scalar scan (O(K^2) on Gram entries)
        if (tid < K_) p[tid] = S[K_ * S_STRIDE + tid];          // x_j . E
        if (tid == 0) aS[0] = S[K_ * S_STRIDE + K_];            // ||E||^2
        __syncthreads();
        for (int r = 0; r < K_; ++r) {
            if (tid == r) {
                float aCur = aS[0];
                float pr   = p[r];
                float srr  = S[r * S_STRIDE + r];
                float dd2  = srr - 2.0f * pr + aCur;            // ||x_r - ema_r||^2
                float ln   = fmaxf(sqrtf(fmaxf(aCur, 0.f)), EPS_);
                surpL[r]   = tanhf(sig_l * sqrtf(fmaxf(dd2, 0.f)) / ln);
                aS[0] = d_l * d_l * aCur + 2.0f * d_l * (1.0f - d_l) * pr
                        + (1.0f - d_l) * (1.0f - d_l) * srr;    // ||ema_{r+1}||^2
            }
            __syncthreads();
            if (tid < K_) p[tid] = d_l * p[tid] + (1.0f - d_l) * S[tid * S_STRIDE + r];
            __syncthreads();
        }

        // (4) per-token gelu reductions + gate_raw (8 tokens/wave, b128 loads)
#pragma unroll
        for (int q = 0; q < 8; ++q) {
            int r = w * 8 + q;
            float dacc = 0.f, nacc = 0.f;
            for (int i = lane * 4; i < D_; i += 128) {
                float4 xv = *(const float4*)(Xc + r * D_ + i);
                float4 ev = *(const float4*)(eo + i);
                float g0 = gelu_f(xv.x), g1 = gelu_f(xv.y);
                float g2 = gelu_f(xv.z), g3 = gelu_f(xv.w);
                dacc += g0 * ev.x + g1 * ev.y + g2 * ev.z + g3 * ev.w;
                nacc += g0 * g0 + g1 * g1 + g2 * g2 + g3 * g3;
            }
#pragma unroll
            for (int off = 16; off; off >>= 1) {
                dacc += __shfl_xor(dacc, off, 32);
                nacc += __shfl_xor(nacc, off, 32);
            }
            if (lane == 0) {
                float cs   = fminf(fmaxf(dacc / fmaxf(sqrtf(nacc), 1e-12f), -1.f), 1.f);
                float gcos = expf(-tau * cs);
                float dg2  = S[r * S_STRIDE + r]
                           - 2.0f * S[(K_ + 1) * S_STRIDE + r]
                           + S[(K_ + 1) * S_STRIDE + (K_ + 1)];
                float sg   = tanhf(sig_g * sqrtf(fmaxf(dg2, 0.f)) / (normG + 2.0f * EPS_));
                gate[b * T_ + c * K_ + r] =
                    gcos * (1.0f + w_g * sg + w_l * surpL[r]);
            }
        }

        // (5) chunk-end EMA: E <- d^64 E + sum_j wgt[j] x_j  (one float4/thread)
        {
            float4* E4 = (float4*)E;
            const float4* X4 = (const float4*)Xc;
            float4 acc = E4[tid];
            acc.x *= dpow64; acc.y *= dpow64; acc.z *= dpow64; acc.w *= dpow64;
#pragma unroll 8
            for (int j = 0; j < K_; ++j) {
                float4 xv = X4[j * (D_ / 4) + tid];
                float wj = wgt[j];
                acc.x += wj * xv.x; acc.y += wj * xv.y;
                acc.z += wj * xv.z; acc.w += wj * xv.w;
            }
            E4[tid] = acc;
        }
    }
}

// ---------------------------------------------------------------------------
// Kernel B: deterministic gate mean (single block)
// ---------------------------------------------------------------------------
__global__ void reduce_gate(const float* __restrict__ gate, float* __restrict__ gmP) {
    __shared__ float red[256];
    float a = 0.f;
    for (int i = threadIdx.x; i < B_ * T_; i += 256) a += gate[i];
    red[threadIdx.x] = a;
    __syncthreads();
    for (int s = 128; s; s >>= 1) {
        if (threadIdx.x < s) red[threadIdx.x] += red[threadIdx.x + s];
        __syncthreads();
    }
    if (threadIdx.x == 0)
        gmP[0] = fmaxf(red[0] / (float)(B_ * T_), EPS_);
}

// ---------------------------------------------------------------------------
// Kernel C: memory-bound finalize, out = gelu(x) * gate/mean (float4 / B128)
// ---------------------------------------------------------------------------
__global__ void __launch_bounds__(256)
finalize_kernel(const float* __restrict__ x, const float* __restrict__ gate,
                const float* __restrict__ gmP, float* __restrict__ out) {
    unsigned long long i4 = (unsigned long long)blockIdx.x * 256ull + threadIdx.x;
    float gm = gmP[0];
    float4 v = ((const float4*)x)[i4];
    unsigned tok = (unsigned)((i4 * 4ull) >> 10);   // D=1024 divides float4 rows
    float s = gate[tok] / gm;
    float4 o;
    o.x = gelu_f(v.x) * s;
    o.y = gelu_f(v.y) * s;
    o.z = gelu_f(v.z) * s;
    o.w = gelu_f(v.w) * s;
    ((float4*)out)[i4] = o;
}

// ---------------------------------------------------------------------------
extern "C" void kernel_launch(void* const* d_in, const int* in_sizes, int n_in,
                              void* d_out, int out_size, void* d_ws, size_t ws_size,
                              hipStream_t stream) {
    const float* x     = (const float*)d_in[0];
    // d_in[1] = logit_decay_g (unused by reference math)
    const float* ldl   = (const float*)d_in[2];
    const float* ltau  = (const float*)d_in[3];
    const float* lsg   = (const float*)d_in[4];
    const float* lsl   = (const float*)d_in[5];
    const float* lwg   = (const float*)d_in[6];
    const float* lwl   = (const float*)d_in[7];
    const float* emaG  = (const float*)d_in[8];
    const float* emaO  = (const float*)d_in[9];
    float* out  = (float*)d_out;
    float* gate = (float*)d_ws;                 // B*T floats
    float* gmP  = gate + B_ * T_;               // 1 float

    gate_kernel<<<B_, 256, SMEM_BYTES, stream>>>(x, ldl, ltau, lsg, lsl, lwg, lwl,
                                                 emaG, emaO, gate);
    reduce_gate<<<1, 256, 0, stream>>>(gate, gmP);
    finalize_kernel<<<(B_ * T_ * D_) / (256 * 4), 256, 0, stream>>>(x, gate, gmP, out);
}
